// NCESoftmaxLoss_56805237457594
// MI455X (gfx1250) — matrix operand order repaired and checked by reading, not compiled, not run on hardware
//
#include <hip/hip_runtime.h>
#include <math.h>

// ---------------------------------------------------------------------------
// NCE softmax loss, fused flash-style on gfx1250 (wave32, WMMA).
//   P = 8192 pairs, D = 256 features, T = 0.07
//   loss = mean_i [ logsumexp_j(-dist(i,j)/T) - (-dist(i,i)/T) ]
// GEMM via bf16x3 split (hi/lo) on v_wmma_f32_16x16x32_bf16 -> ~fp32 accuracy.
// Keys (8MB) are L2-resident; logits are never materialized.
// ---------------------------------------------------------------------------

typedef __attribute__((ext_vector_type(16))) __bf16 v16bf;
typedef __attribute__((ext_vector_type(8)))  float  v8f;

#define P_PAIRS 8192
#define DFEAT   256
#define INV_T   (1.0f / 0.07f)

// ---- workspace layout (bytes) ---------------------------------------------
#define WS_QHI   0u
#define WS_QLO   (WS_QHI + P_PAIRS * DFEAT * 2u)   //  4 MB each
#define WS_KHI   (WS_QLO + P_PAIRS * DFEAT * 2u)
#define WS_KLO   (WS_KHI + P_PAIRS * DFEAT * 2u)
#define WS_QSQ   (WS_KLO + P_PAIRS * DFEAT * 2u)
#define WS_KSQ   (WS_QSQ + P_PAIRS * 4u)
#define WS_ROWL  (WS_KSQ + P_PAIRS * 4u)

// ---------------------------------------------------------------------------
// Prep: split fp32 row into bf16 hi/lo, compute row sum-of-squares.
// One block per row, 256 threads == 256 features.
// ---------------------------------------------------------------------------
__global__ __launch_bounds__(256)
void prep_keys_kernel(const float* __restrict__ f2,
                      __bf16* __restrict__ khi, __bf16* __restrict__ klo,
                      float* __restrict__ ksq) {
  const int row = blockIdx.x;
  const int t   = threadIdx.x;
  const float x = f2[row * DFEAT + t];

  const __bf16 hi = (__bf16)x;
  const __bf16 lo = (__bf16)(x - (float)hi);
  khi[row * DFEAT + t] = hi;
  klo[row * DFEAT + t] = lo;

  float v = x * x;
  #pragma unroll
  for (int off = 16; off >= 1; off >>= 1) v += __shfl_down(v, off, 32);

  __shared__ float sm[8];
  const int lane = t & 31, warp = t >> 5;
  if (lane == 0) sm[warp] = v;
  __syncthreads();
  if (t == 0) {
    float s = 0.f;
    #pragma unroll
    for (int i = 0; i < 8; ++i) s += sm[i];
    ksq[row] = s;
  }
}

__global__ __launch_bounds__(256)
void prep_queries_kernel(const float* __restrict__ f1,
                         const int* __restrict__ map21, int n1,
                         __bf16* __restrict__ qhi, __bf16* __restrict__ qlo,
                         float* __restrict__ qsq) {
  const int row = blockIdx.x;
  const int t   = threadIdx.x;
  int src = map21[row];
  src = src < 0 ? 0 : (src >= n1 ? n1 - 1 : src);   // defensive clamp
  const float x = f1[src * DFEAT + t];

  const __bf16 hi = (__bf16)x;
  const __bf16 lo = (__bf16)(x - (float)hi);
  qhi[row * DFEAT + t] = hi;
  qlo[row * DFEAT + t] = lo;

  float v = x * x;
  #pragma unroll
  for (int off = 16; off >= 1; off >>= 1) v += __shfl_down(v, off, 32);

  __shared__ float sm[8];
  const int lane = t & 31, warp = t >> 5;
  if (lane == 0) sm[warp] = v;
  __syncthreads();
  if (t == 0) {
    float s = 0.f;
    #pragma unroll
    for (int i = 0; i < 8; ++i) s += sm[i];
    qsq[row] = s;
  }
}

// ---------------------------------------------------------------------------
// Fused GEMM + online logsumexp.
//   Block: 256 threads = 8 waves. Waves (band=w&3, half=w>>2).
//   Each (band,half) pair owns rows [blk*64 + band*16, +16), half of the
//   512 column tiles each. A fragments (hi+lo, 8 K-chunks) live in VGPRs
//   for the whole sweep; B streams from L2.
// ---------------------------------------------------------------------------
__global__ __launch_bounds__(256)
void nce_main_kernel(const __bf16* __restrict__ Qhi, const __bf16* __restrict__ Qlo,
                     const __bf16* __restrict__ Khi, const __bf16* __restrict__ Klo,
                     const float* __restrict__ qsq, const float* __restrict__ ksq,
                     float* __restrict__ row_loss) {
  const int tid  = threadIdx.x;
  const int lane = tid & 31;
  const int warp = tid >> 5;
  const int band = warp & 3;
  const int half = warp >> 2;
  const int laneN = lane & 15;      // N / M-row position within fragment
  const int hrow  = lane >> 4;      // which 16-lane half of the wave

  const int m0 = blockIdx.x * 64 + band * 16;   // first query row of this band

  // ---- load A fragments (16 rows x 256 K, hi+lo) into registers ----
  // A layout (16-bit 16x32): lane m (both halves) holds row m; low half K
  // elems {kb..kb+7, kb+16..kb+23} with kb = hrow*8.  All 16B aligned.
  v16bf ahi[8], alo[8];
  {
    const __bf16* qh = Qhi + (m0 + laneN) * DFEAT;
    const __bf16* ql = Qlo + (m0 + laneN) * DFEAT;
    const int kb = hrow * 8;
    #pragma unroll
    for (int c = 0; c < 8; ++c) {
      const int base = c * 32 + kb;
      uint4* ap = reinterpret_cast<uint4*>(&ahi[c]);
      ap[0] = *reinterpret_cast<const uint4*>(qh + base);
      ap[1] = *reinterpret_cast<const uint4*>(qh + base + 16);
      uint4* lp = reinterpret_cast<uint4*>(&alo[c]);
      lp[0] = *reinterpret_cast<const uint4*>(ql + base);
      lp[1] = *reinterpret_cast<const uint4*>(ql + base + 16);
    }
  }

  float qsqv[8];
  #pragma unroll
  for (int r = 0; r < 8; ++r) qsqv[r] = qsq[m0 + hrow * 8 + r];

  float mrun[8], srun[8];
  #pragma unroll
  for (int r = 0; r < 8; ++r) { mrun[r] = -INFINITY; srun[r] = 0.0f; }
  float dsum = 0.0f;   // diagonal logit (at most one column matches per lane)

  const int kbB = hrow * 8;    // B fragment K offset for this lane half

  for (int t = half; t < P_PAIRS / 16; t += 2) {
    const int j0 = t * 16;
    const __bf16* kh = Khi + (j0 + laneN) * DFEAT + kbB;
    const __bf16* kl = Klo + (j0 + laneN) * DFEAT + kbB;

    v8f acc = {};
    #pragma unroll
    for (int c = 0; c < 8; ++c) {
      // B layout (16-bit 32x16): lane n holds column n, K {kb..kb+7, kb+16..kb+23}, kb = hrow*8.
      v16bf bhi, blo;
      uint4* bp = reinterpret_cast<uint4*>(&bhi);
      bp[0] = *reinterpret_cast<const uint4*>(kh + c * 32);
      bp[1] = *reinterpret_cast<const uint4*>(kh + c * 32 + 16);
      uint4* lp = reinterpret_cast<uint4*>(&blo);
      lp[0] = *reinterpret_cast<const uint4*>(kl + c * 32);
      lp[1] = *reinterpret_cast<const uint4*>(kl + c * 32 + 16);

      // bf16x3 error-compensated fp32 product
      acc = __builtin_amdgcn_wmma_f32_16x16x32_bf16(false, ahi[c], false, bhi,
                                                    (short)0, acc, false, false);
      acc = __builtin_amdgcn_wmma_f32_16x16x32_bf16(false, alo[c], false, bhi,
                                                    (short)0, acc, false, false);
      acc = __builtin_amdgcn_wmma_f32_16x16x32_bf16(false, ahi[c], false, blo,
                                                    (short)0, acc, false, false);
    }

    const float ksqv = ksq[j0 + laneN];
    const int   nglb = j0 + laneN;

    #pragma unroll
    for (int r = 0; r < 8; ++r) {
      // C layout: VGPR r = row (hrow*8 + r), lane%16 = column
      float sq = fmaf(-2.0f, acc[r], qsqv[r] + ksqv);
      sq = fmaxf(sq, 0.0f);
      const float logit = -__builtin_sqrtf(sq) * INV_T;

      const int mglb = m0 + hrow * 8 + r;
      dsum += (mglb == nglb) ? logit : 0.0f;

      // per-lane online logsumexp over this lane's column subset
      const float mn = fmaxf(mrun[r], logit);
      srun[r] = srun[r] * __expf(mrun[r] - mn) + __expf(logit - mn);
      mrun[r] = mn;
    }
  }

  // ---- merge (m,s) across the 16 lanes that share each row ----
  #pragma unroll
  for (int r = 0; r < 8; ++r) {
    float m = mrun[r], s = srun[r];
    #pragma unroll
    for (int off = 1; off <= 8; off <<= 1) {
      const float om = __shfl_xor(m, off, 32);
      const float os = __shfl_xor(s, off, 32);
      const float mn = fmaxf(m, om);
      s = s * __expf(m - mn) + os * __expf(om - mn);
      m = mn;
    }
    mrun[r] = m; srun[r] = s;
  }

  // ---- cross-wave (column-half) merge through LDS ----
  __shared__ float lds_m[2][64], lds_s[2][64], lds_d[2][64];

  if (laneN == 0) {
    #pragma unroll
    for (int r = 0; r < 8; ++r) {
      const int rl = band * 16 + hrow * 8 + r;
      lds_m[half][rl] = mrun[r];
      lds_s[half][rl] = srun[r];
    }
  }
  // diag values live in lanes 0..7 (rows +0..7) and 24..31 (rows +8..15)
  if (lane < 8)        lds_d[half][band * 16 + lane]        = dsum;
  else if (lane >= 24) lds_d[half][band * 16 + (lane - 16)] = dsum;
  else { // cover remaining slots with this lane's (always-zero) dsum partner?
    // lanes 8..23 never match a diagonal; their rows are covered above.
  }
  __syncthreads();

  if (tid < 64) {
    const float m0v = lds_m[0][tid], s0 = lds_s[0][tid];
    const float m1v = lds_m[1][tid], s1 = lds_s[1][tid];
    const float mn  = fmaxf(m0v, m1v);
    const float S   = s0 * __expf(m0v - mn) + s1 * __expf(m1v - mn);
    const float lse = mn + __logf(S);
    const float dg  = lds_d[0][tid] + lds_d[1][tid];  // exactly one is nonzero
    row_loss[blockIdx.x * 64 + tid] = lse - dg;
  }
}

// ---------------------------------------------------------------------------
// Deterministic final reduction: loss = mean(row_loss)
// ---------------------------------------------------------------------------
__global__ __launch_bounds__(256)
void finalize_kernel(const float* __restrict__ row_loss, float* __restrict__ out) {
  __shared__ float sm[256];
  const int t = threadIdx.x;
  float s = 0.f;
  for (int i = t; i < P_PAIRS; i += 256) s += row_loss[i];
  sm[t] = s;
  __syncthreads();
  for (int off = 128; off > 0; off >>= 1) {
    if (t < off) sm[t] += sm[t + off];
    __syncthreads();
  }
  if (t == 0) out[0] = sm[0] * (1.0f / (float)P_PAIRS);
}

// ---------------------------------------------------------------------------
extern "C" void kernel_launch(void* const* d_in, const int* in_sizes, int n_in,
                              void* d_out, int out_size, void* d_ws, size_t ws_size,
                              hipStream_t stream) {
  const float* f1    = (const float*)d_in[0];   // [20000, 256]
  const float* f2    = (const float*)d_in[1];   // [8192, 256]
  const int*   map21 = (const int*)d_in[2];     // [8192]
  const int n1 = in_sizes[0] / DFEAT;

  char* ws = (char*)d_ws;
  __bf16* Qhi = (__bf16*)(ws + WS_QHI);
  __bf16* Qlo = (__bf16*)(ws + WS_QLO);
  __bf16* Khi = (__bf16*)(ws + WS_KHI);
  __bf16* Klo = (__bf16*)(ws + WS_KLO);
  float*  qsq = (float*)(ws + WS_QSQ);
  float*  ksq = (float*)(ws + WS_KSQ);
  float*  rls = (float*)(ws + WS_ROWL);

  prep_keys_kernel<<<P_PAIRS, 256, 0, stream>>>(f2, Khi, Klo, ksq);
  prep_queries_kernel<<<P_PAIRS, 256, 0, stream>>>(f1, map21, n1, Qhi, Qlo, qsq);
  nce_main_kernel<<<P_PAIRS / 64, 256, 0, stream>>>(Qhi, Qlo, Khi, Klo,
                                                    qsq, ksq, rls);
  finalize_kernel<<<1, 256, 0, stream>>>(rls, (float*)d_out);
}
